// MaxPoolingAggregator_76166950028261
// MI455X (gfx1250) — matrix operand order repaired and checked by reading, not compiled, not run on hardware
//
#include <hip/hip_runtime.h>
#include <stdint.h>

typedef __attribute__((ext_vector_type(16))) __bf16 v16bf;
typedef __attribute__((ext_vector_type(8)))  float  v8f;

union FragU { uint4 u[2]; v16bf v; };

__device__ __forceinline__ unsigned short f2bf(float f) {
    unsigned u = __float_as_uint(f);
    u = u + 0x7FFFu + ((u >> 16) & 1u);   // round-to-nearest-even
    return (unsigned short)(u >> 16);
}
__device__ __forceinline__ unsigned pack2(float lo, float hi) {
    return (unsigned)f2bf(lo) | ((unsigned)f2bf(hi) << 16);
}

// ---------------------------------------------------------------------------
// Pack a row-major fp32 weight matrix W[outd][ind] into bf16 WMMA-B fragments:
// dst dword index = ((ks*NT + nt)*32 + lane)*8 + v
//   n  = nt*16 + (lane & 15)
//   k0 = ks*32 + (lane >> 4)*8 + 2*(v&3) + 16*(v>>2)  (B 32x16: lanes 0-15 K=0..7,16..23,
//                                            lanes 16-31 K=8..15,24..31, 2 K / VGPR)
// ---------------------------------------------------------------------------
#define NT 11  // 176 padded output columns

__global__ void pack_kernel(const float* __restrict__ W, int outd, int ind,
                            int KT, unsigned* __restrict__ dst) {
    int idx = blockIdx.x * blockDim.x + threadIdx.x;
    int total = KT * NT * 32 * 8;
    if (idx >= total) return;
    int v    = idx & 7;
    int lane = (idx >> 3) & 31;
    int nt   = (idx >> 8) % NT;
    int ks   = idx / (NT * 256);
    int n  = nt * 16 + (lane & 15);
    int k0 = ks * 32 + (lane >> 4) * 8 + 2 * (v & 3) + 16 * (v >> 2);
    float lo = (n < outd && k0     < ind) ? W[(size_t)n * ind + k0]     : 0.f;
    float hi = (n < outd && k0 + 1 < ind) ? W[(size_t)n * ind + k0 + 1] : 0.f;
    dst[idx] = pack2(lo, hi);
}

// ---------------------------------------------------------------------------
// Aggregation: per batch element b (8192 workgroups, 352 threads = 11 waves)
//   h[n,:] = relu((concat(nbr,edge,time)[n] @ Wt^T + b_t) * w[n] * valid[n])
//   pooled = max over n
// M = 64 (50 rows padded), K = 448 (444 padded), Ncols = 176 (172 padded)
// ---------------------------------------------------------------------------
#define A_STRIDE 456            // 448 + 8 pad (bf16 elems), 912B row = 16B mult

__global__ __launch_bounds__(352) void agg_kernel(
    const float* __restrict__ nbrF, const float* __restrict__ edgeF,
    const float* __restrict__ timeF, const unsigned char* __restrict__ mask,
    const float* __restrict__ wts, const float* __restrict__ b_t,
    const unsigned* __restrict__ Wt_pk, float* __restrict__ pooled) {

    __shared__ unsigned short sA[64 * A_STRIDE];
    __shared__ float sW[64];

    const int b    = blockIdx.x;
    const int tid  = threadIdx.x;
    const int lane = tid & 31;
    const int wid  = tid >> 5;          // 0..10 == n-tile

    if (tid < 64) {
        float w = 0.f;
        if (tid < 50 && mask[(size_t)b * 50 + tid] == 0)
            w = wts[(size_t)b * 50 + tid];
        sW[tid] = w;
    }

    // Stage A: concat [172|172|100] fp32 -> bf16 LDS, zero-padded to 64x456.
    for (int r = wid; r < 64; r += 11) {
        const size_t rowbase = (size_t)b * 50 + r;
#pragma unroll
        for (int it = 0; it < 4; ++it) {
            int c4 = lane + 32 * it;            // float4 slot
            if (c4 >= 114) continue;            // 456/4 slots per row
            int col = c4 * 4;
            float4 v = make_float4(0.f, 0.f, 0.f, 0.f);
            if (r < 50) {
                if (col < 172)
                    v = *(const float4*)(nbrF + rowbase * 172 + col);
                else if (col < 344)
                    v = *(const float4*)(edgeF + rowbase * 172 + (col - 172));
                else if (col < 444)
                    v = *(const float4*)(timeF + rowbase * 100 + (col - 344));
            }
            uint2 p = make_uint2(pack2(v.x, v.y), pack2(v.z, v.w));
            *(uint2*)(&sA[r * A_STRIDE + col]) = p;
        }
    }
    __syncthreads();

    const int nt = wid;
    v8f acc[4] = {};
    const int row0  = lane & 15;
    const int khalf = (lane >> 4) * 8;

    for (int ks = 0; ks < 14; ++ks) {
        FragU bf;
        const uint4* bp =
            (const uint4*)(Wt_pk + (((size_t)(ks * NT + nt) * 32 + lane) * 8));
        bf.u[0] = bp[0];
        bf.u[1] = bp[1];
        const int basek = ks * 32 + khalf;
#pragma unroll
        for (int mt = 0; mt < 4; ++mt) {
            FragU af;
            const unsigned short* ap = &sA[(mt * 16 + row0) * A_STRIDE + basek];
            af.u[0] = *(const uint4*)ap;          // K chunk +0..7
            af.u[1] = *(const uint4*)(ap + 16);   // K chunk +16..23
            acc[mt] = __builtin_amdgcn_wmma_f32_16x16x32_bf16(
                false, af.v, false, bf.v, (short)0, acc[mt], false, false);
        }
    }

    // Fused epilogue: + b_t, scale, relu, row-max.
    const int col = nt * 16 + (lane & 15);
    const float bt = (col < 172) ? b_t[col] : 0.f;
    const int roff = (lane >> 4) * 8;
    float pm = 0.f;
#pragma unroll
    for (int mt = 0; mt < 4; ++mt) {
#pragma unroll
        for (int r = 0; r < 8; ++r) {
            int row = mt * 16 + roff + r;
            float v = (acc[mt][r] + bt) * sW[row];
            pm = fmaxf(pm, fmaxf(v, 0.f));
        }
    }
    pm = fmaxf(pm, __shfl_xor(pm, 16, 32));       // merge lane halves
    if (lane < 16 && col < 172)
        pooled[(size_t)b * 172 + col] = pm;
}

// ---------------------------------------------------------------------------
// Merge MLP: out = W2 @ relu(W1 @ [pooled|src] + b1) + b2
// 16 batch rows / workgroup, 11 waves (wave = n-tile), fc1 K=352, fc2 K=192.
// ---------------------------------------------------------------------------
#define A2_STRIDE 360   // 352 + 8 pad
#define H_STRIDE  200   // 192 + 8 pad

__global__ __launch_bounds__(352) void merge_kernel(
    const float* __restrict__ pooled, const float* __restrict__ srcF,
    const float* __restrict__ b1v, const float* __restrict__ b2v,
    const unsigned* __restrict__ W1_pk, const unsigned* __restrict__ W2_pk,
    float* __restrict__ out) {

    __shared__ unsigned short sA[16 * A2_STRIDE];
    __shared__ unsigned short sH[16 * H_STRIDE];

    const int b0   = blockIdx.x * 16;
    const int tid  = threadIdx.x;
    const int lane = tid & 31;
    const int nt   = tid >> 5;

    for (int i = tid; i < 16 * H_STRIDE; i += 352) sH[i] = 0;

    // Stage concat [pooled(172) | src(172) | pad(8)] -> bf16 LDS.
    for (int q = tid; q < 16 * 90; q += 352) {      // 360/4 = 90 slots
        int r = q / 90, c4 = q % 90, col = c4 * 4;
        float4 v = make_float4(0.f, 0.f, 0.f, 0.f);
        if (col < 172)
            v = *(const float4*)(pooled + (size_t)(b0 + r) * 172 + col);
        else if (col < 344)
            v = *(const float4*)(srcF + (size_t)(b0 + r) * 172 + (col - 172));
        uint2 p = make_uint2(pack2(v.x, v.y), pack2(v.z, v.w));
        *(uint2*)(&sA[r * A2_STRIDE + col]) = p;
    }
    __syncthreads();

    const int row0  = lane & 15;
    const int khalf = (lane >> 4) * 8;
    const int col   = nt * 16 + (lane & 15);
    const int roff  = (lane >> 4) * 8;

    // fc1
    v8f acc = {};
    for (int ks = 0; ks < 11; ++ks) {
        FragU bf;
        const uint4* bp =
            (const uint4*)(W1_pk + (((size_t)(ks * NT + nt) * 32 + lane) * 8));
        bf.u[0] = bp[0];
        bf.u[1] = bp[1];
        FragU af;
        const unsigned short* ap = &sA[row0 * A2_STRIDE + ks * 32 + khalf];
        af.u[0] = *(const uint4*)ap;
        af.u[1] = *(const uint4*)(ap + 16);
        acc = __builtin_amdgcn_wmma_f32_16x16x32_bf16(
            false, af.v, false, bf.v, (short)0, acc, false, false);
    }
    const float bb1 = (col < 172) ? b1v[col] : 0.f;
#pragma unroll
    for (int r = 0; r < 8; ++r) {
        float h = fmaxf(acc[r] + bb1, 0.f);
        sH[(roff + r) * H_STRIDE + col] = f2bf(h);
    }
    __syncthreads();

    // fc2
    v8f acc2 = {};
    for (int ks = 0; ks < 6; ++ks) {
        FragU bf;
        const uint4* bp =
            (const uint4*)(W2_pk + (((size_t)(ks * NT + nt) * 32 + lane) * 8));
        bf.u[0] = bp[0];
        bf.u[1] = bp[1];
        FragU af;
        const unsigned short* ap = &sH[row0 * H_STRIDE + ks * 32 + khalf];
        af.u[0] = *(const uint4*)ap;
        af.u[1] = *(const uint4*)(ap + 16);
        acc2 = __builtin_amdgcn_wmma_f32_16x16x32_bf16(
            false, af.v, false, bf.v, (short)0, acc2, false, false);
    }
    if (col < 172) {
        const float bb2 = b2v[col];
#pragma unroll
        for (int r = 0; r < 8; ++r)
            out[(size_t)(b0 + roff + r) * 172 + col] = acc2[r] + bb2;
    }
}

// ---------------------------------------------------------------------------
extern "C" void kernel_launch(void* const* d_in, const int* in_sizes, int n_in,
                              void* d_out, int out_size, void* d_ws, size_t ws_size,
                              hipStream_t stream) {
    (void)in_sizes; (void)n_in; (void)out_size; (void)ws_size;
    const float* srcF  = (const float*)d_in[0];
    const float* nbrF  = (const float*)d_in[1];
    const float* timeF = (const float*)d_in[2];
    const float* edgeF = (const float*)d_in[3];
    const unsigned char* mask = (const unsigned char*)d_in[4];
    const float* wts   = (const float*)d_in[5];
    const float* W_t   = (const float*)d_in[6];
    const float* b_t   = (const float*)d_in[7];
    const float* W1    = (const float*)d_in[8];
    const float* b1v   = (const float*)d_in[9];
    const float* W2    = (const float*)d_in[10];
    const float* b2v   = (const float*)d_in[11];
    float* out = (float*)d_out;

    // Workspace layout (dwords)
    const size_t WT_DW = (size_t)14 * NT * 32 * 8;  // 39424
    const size_t W1_DW = (size_t)11 * NT * 32 * 8;  // 30976
    const size_t W2_DW = (size_t)6  * NT * 32 * 8;  // 16896
    unsigned* wt_pk = (unsigned*)d_ws;
    unsigned* w1_pk = wt_pk + WT_DW;
    unsigned* w2_pk = w1_pk + W1_DW;
    float*    pool  = (float*)(w2_pk + W2_DW);      // 8192*172 floats

    pack_kernel<<<(int)((WT_DW + 255) / 256), 256, 0, stream>>>(W_t, 172, 444, 14, wt_pk);
    pack_kernel<<<(int)((W1_DW + 255) / 256), 256, 0, stream>>>(W1, 172, 344, 11, w1_pk);
    pack_kernel<<<(int)((W2_DW + 255) / 256), 256, 0, stream>>>(W2, 172, 172, 6, w2_pk);

    agg_kernel<<<8192, 352, 0, stream>>>(nbrF, edgeF, timeF, mask, wts, b_t,
                                         wt_pk, pool);
    merge_kernel<<<512, 352, 0, stream>>>(pool, srcF, b1v, b2v, w1_pk, w2_pk,
                                          out);
}
